// Physics_Attention_1D_Eidetic_11106785428039
// MI455X (gfx1250) — matrix operand order repaired and checked by reading, not compiled, not run on hardware
//
#include <hip/hip_runtime.h>
#include <hip/hip_bf16.h>
#include <math.h>

typedef __attribute__((ext_vector_type(16))) _Float16 v16h;
typedef __attribute__((ext_vector_type(8)))  _Float16 v8h;
typedef __attribute__((ext_vector_type(8)))  float    v8f;

#define WMMA_F16(a,b,c) __builtin_amdgcn_wmma_f32_16x16x32_f16(false,(a),false,(b),(short)0,(c),false,false)

constexpr int Bb = 4, Ntok = 16384, Dm = 256, Hh = 8, DH = 64, Gs = 64, INNER = 512;

__device__ __forceinline__ float gelu_exact(float x) {
    return 0.5f * x * (1.0f + erff(x * 0.70710678118654752f));
}

// --- CDNA5 async global->LDS copy (16B per lane), tracked by ASYNCcnt ---
// LDS operand: low 32 bits of the shared-aperture generic address == wave-relative LDS address.
__device__ __forceinline__ void async_ld_b128(void* lds_dst, const void* gsrc) {
    unsigned loff = (unsigned)(uintptr_t)lds_dst;
    unsigned long long ga = (unsigned long long)(uintptr_t)gsrc;
    asm volatile("global_load_async_to_lds_b128 %0, %1, off" :: "v"(loff), "v"(ga) : "memory");
}
__device__ __forceinline__ void wait_async0() {
    asm volatile("s_wait_asynccnt 0" ::: "memory");
}

// A fragment 16x32 f16: lane holds row m=lane&15; K = koff..koff+7 and koff+16..koff+23,
// koff = 8*(lane>=16). base = tile origin (row-major, stride ld halves, 16B-aligned rows).
__device__ __forceinline__ v16h frag_a(const _Float16* base, int lane, int ld) {
    const _Float16* p = base + (lane & 15) * ld + ((lane >> 4) << 3);
    v8h lo = *(const v8h*)p;
    v8h hi = *(const v8h*)(p + 16);
    v16h a;
#pragma unroll
    for (int i = 0; i < 8; i++) { a[i] = lo[i]; a[i + 8] = hi[i]; }
    return a;
}

// B fragment 32x16 f16 from LDS tile stored TRANSPOSED (Bt[n][k]): lane holds col n=lane&15,
// K = koff..koff+7 and koff+16..koff+23, koff = 8*(lane>=16).
__device__ __forceinline__ v16h frag_b(const _Float16* baseT, int lane, int ld) {
    const _Float16* p = baseT + (lane & 15) * ld + ((lane >> 4) << 3);
    v8h lo = *(const v8h*)p;
    v8h hi = *(const v8h*)(p + 16);
    v16h b;
#pragma unroll
    for (int i = 0; i < 8; i++) { b[i] = lo[i]; b[i + 8] = hi[i]; }
    return b;
}

// ---------------- K1: x_mid = x @ Wx^T + bx   (M=65536, N=512, K=256) ----------------
// 64x64 block tile, 8 waves, 2 N-tiles per wave sharing one A fragment.
__global__ __launch_bounds__(256) void k1_proj_x(const float* __restrict__ x,
                                                 const float* __restrict__ Wx,
                                                 const float* __restrict__ bx,
                                                 _Float16* __restrict__ xmid) {
    __shared__ alignas(16) _Float16 As[64 * 40];
    __shared__ alignas(16) _Float16 Bs[64 * 40];
    const int tid = threadIdx.x, lane = tid & 31, wid = tid >> 5;
    const int mblk = blockIdx.x, nblk = blockIdx.y;
    const int wm = wid & 3, wn0 = (wid >> 2) << 1;
    const int sr = tid >> 2, sc8 = (tid & 3) * 8;   // staging coords (64 rows x 8 cols each)
    v8f acc0 = {}, acc1 = {};
    for (int k0 = 0; k0 < Dm; k0 += 32) {
        {   // stage A tile 64x32 (f32 -> f16)
            const float* src = x + (size_t)(mblk * 64 + sr) * Dm + k0 + sc8;
            float4 v0 = *(const float4*)(src);
            float4 v1 = *(const float4*)(src + 4);
            _Float16* p = As + sr * 40 + sc8;
            p[0] = (_Float16)v0.x; p[1] = (_Float16)v0.y; p[2] = (_Float16)v0.z; p[3] = (_Float16)v0.w;
            p[4] = (_Float16)v1.x; p[5] = (_Float16)v1.y; p[6] = (_Float16)v1.z; p[7] = (_Float16)v1.w;
            if (k0 + 32 < Dm) __builtin_prefetch(src + 32, 0, 0);
        }
        {   // stage B^T tile 64x32: Bt[n][k] = Wx[n][k]
            const float* src = Wx + (size_t)(nblk * 64 + sr) * Dm + k0 + sc8;
            float4 v0 = *(const float4*)(src);
            float4 v1 = *(const float4*)(src + 4);
            _Float16* p = Bs + sr * 40 + sc8;
            p[0] = (_Float16)v0.x; p[1] = (_Float16)v0.y; p[2] = (_Float16)v0.z; p[3] = (_Float16)v0.w;
            p[4] = (_Float16)v1.x; p[5] = (_Float16)v1.y; p[6] = (_Float16)v1.z; p[7] = (_Float16)v1.w;
            if (k0 + 32 < Dm) __builtin_prefetch(src + 32, 0, 0);
        }
        __syncthreads();
        v16h a  = frag_a(As + wm * 16 * 40, lane, 40);
        v16h b0 = frag_b(Bs + wn0 * 16 * 40, lane, 40);
        v16h b1 = frag_b(Bs + (wn0 + 1) * 16 * 40, lane, 40);
        acc0 = WMMA_F16(a, b0, acc0);
        acc1 = WMMA_F16(a, b1, acc1);
        __syncthreads();
    }
    const int mbase = mblk * 64 + wm * 16 + ((lane >> 4) << 3);
    const int n0 = nblk * 64 + wn0 * 16 + (lane & 15);
    const int n1 = n0 + 16;
    const float bn0 = bx[n0], bn1 = bx[n1];
    const int h0 = n0 >> 6, c0 = n0 & 63, h1 = n1 >> 6, c1 = n1 & 63;
#pragma unroll
    for (int r = 0; r < 8; r++) {
        int m = mbase + r;
        int b_ = m >> 14, tok = m & 16383;
        xmid[((size_t)(b_ * Hh + h0) * Ntok + tok) * DH + c0] = (_Float16)(acc0[r] + bn0);
        xmid[((size_t)(b_ * Hh + h1) * Ntok + tok) * DH + c1] = (_Float16)(acc1[r] + bn1);
    }
}

// ---------------- K2: routing (temp MLP + gumbel softmax) per 32-token tile ----------------
__global__ __launch_bounds__(256) void k2_route(const _Float16* __restrict__ xmid,
                                                const float* __restrict__ W1, const float* __restrict__ b1,
                                                const float* __restrict__ W2, const float* __restrict__ b2,
                                                const float* __restrict__ biasH,
                                                const float* __restrict__ Wsl, const float* __restrict__ bsl,
                                                const float* __restrict__ u,
                                                _Float16* __restrict__ sw, float* __restrict__ norm_g) {
    __shared__ alignas(16) _Float16 W1t[64 * 72];
    __shared__ alignas(16) _Float16 Wst[64 * 72];
    __shared__ alignas(16) _Float16 At[32 * 72];
    __shared__ float Hs[32 * 64];
    __shared__ float Ls[32 * 64];
    const int tid = threadIdx.x, lane = tid & 31, wid = tid >> 5;
    const int tchunk = blockIdx.x, bh = blockIdx.y;
    const int h = bh & 7;
    const int tok0 = tchunk * 32;
    {   // stage x_mid tile 32x64 via CDNA5 async global->LDS (straight f16 copy, 16B/lane)
        int r = tid >> 3, c8 = (tid & 7) * 8;
        async_ld_b128(At + r * 72 + c8, xmid + ((size_t)bh * Ntok + tok0 + r) * DH + c8);
    }
    {   // stage both weight matrices (f32 -> f16, row n holds K contiguous)
        int n = tid >> 2, k16 = (tid & 3) * 16;
#pragma unroll
        for (int i = 0; i < 16; i++) {
            W1t[n * 72 + k16 + i] = (_Float16)W1[n * 64 + k16 + i];
            Wst[n * 72 + k16 + i] = (_Float16)Wsl[n * 64 + k16 + i];
        }
    }
    wait_async0();
    __syncthreads();
    const int wm = wid & 1, wn = wid >> 1;
    const int mrow = wm * 16 + ((lane >> 4) << 3);
    const int ncol = wn * 16 + (lane & 15);
    {   // hidden = gelu(x_mid @ W1^T + b1)
        v8f acc = {};
#pragma unroll
        for (int ks = 0; ks < 64; ks += 32) {
            v16h a = frag_a(At + wm * 16 * 72 + ks, lane, 72);
            v16h b = frag_b(W1t + wn * 16 * 72 + ks, lane, 72);
            acc = WMMA_F16(a, b, acc);
        }
        float bn = b1[ncol];
#pragma unroll
        for (int r = 0; r < 8; r++) Hs[(mrow + r) * 64 + ncol] = gelu_exact(acc[r] + bn);
    }
    {   // logits = x_mid @ Ws^T + bs
        v8f acc = {};
#pragma unroll
        for (int ks = 0; ks < 64; ks += 32) {
            v16h a = frag_a(At + wm * 16 * 72 + ks, lane, 72);
            v16h b = frag_b(Wst + wn * 16 * 72 + ks, lane, 72);
            acc = WMMA_F16(a, b, acc);
        }
        float bn = bsl[ncol];
#pragma unroll
        for (int r = 0; r < 8; r++) Ls[(mrow + r) * 64 + ncol] = acc[r] + bn;
    }
    __syncthreads();
    if (tid < 32) {   // per-token temperature + gumbel softmax (token = tid)
        const int t = tid;
        float accum = b2[0];
        for (int g = 0; g < 64; g++) accum += Hs[t * 64 + g] * W2[g];
        float temp = gelu_exact(accum) + biasH[h];
        temp = fmaxf(temp, 0.01f);
        const float invT = 1.0f / temp;
        const size_t ub = ((size_t)bh * Ntok + tok0 + t) * (size_t)Gs;
        float mx = -1e30f;
        for (int g = 0; g < 64; g++) {
            float uv = u[ub + g];
            float gn = -logf(-logf(uv + 1e-8f) + 1e-8f);
            float v = (Ls[t * 64 + g] + gn) * invT;
            Ls[t * 64 + g] = v;
            mx = fmaxf(mx, v);
        }
        float ssum = 0.0f;
        for (int g = 0; g < 64; g++) {
            float e = expf(Ls[t * 64 + g] - mx);
            Ls[t * 64 + g] = e;
            ssum += e;
        }
        float inv = 1.0f / ssum;
        _Float16* dst = sw + ub;
        for (int g = 0; g < 64; g++) {
            float w = Ls[t * 64 + g] * inv;
            Ls[t * 64 + g] = w;
            dst[g] = (_Float16)w;
        }
    }
    __syncthreads();
    if (tid < 64) {   // block-partial slice_norm
        int g = tid;
        float s = 0.0f;
        for (int r = 0; r < 32; r++) s += Ls[r * 64 + g];
        atomicAdd(&norm_g[bh * 64 + g], s);
    }
}

// ---------------- K3: slice_tok[g,c] = sum_n slice_w[n,g] * x_mid[n,c] ----------------
__global__ __launch_bounds__(256) void k3_pool(const _Float16* __restrict__ sw,
                                               const _Float16* __restrict__ xmid,
                                               float* __restrict__ stok) {
    __shared__ alignas(16) _Float16 As[64 * 40];   // [g][k]
    __shared__ alignas(16) _Float16 Bt[64 * 40];   // [c][k]
    const int tid = threadIdx.x, lane = tid & 31, wid = tid >> 5;
    const int ksplit = blockIdx.x, bh = blockIdx.y;
    const int wm = wid >> 1, wn0 = (wid & 1) * 2;
    v8f acc0 = {}, acc1 = {};
    for (int step = 0; step < 16; step++) {
        const int tok0 = ksplit * 512 + step * 32;
        {   // transpose-stage both tiles: K dim = tokens
            int kk = tid >> 3, q8 = (tid & 7) * 8;
            const _Float16* srcA = sw   + ((size_t)bh * Ntok + tok0 + kk) * Gs + q8;
            const _Float16* srcB = xmid + ((size_t)bh * Ntok + tok0 + kk) * DH + q8;
#pragma unroll
            for (int i = 0; i < 8; i++) {
                As[(q8 + i) * 40 + kk] = srcA[i];
                Bt[(q8 + i) * 40 + kk] = srcB[i];
            }
        }
        __syncthreads();
        v16h a  = frag_a(As + wm * 16 * 40, lane, 40);
        v16h b0 = frag_b(Bt + wn0 * 16 * 40, lane, 40);
        v16h b1 = frag_b(Bt + (wn0 + 1) * 16 * 40, lane, 40);
        acc0 = WMMA_F16(a, b0, acc0);
        acc1 = WMMA_F16(a, b1, acc1);
        __syncthreads();
    }
    const int gbase = wm * 16 + ((lane >> 4) << 3);
    const int c0 = wn0 * 16 + (lane & 15);
#pragma unroll
    for (int r = 0; r < 8; r++) {
        atomicAdd(&stok[(size_t)bh * 4096 + (gbase + r) * 64 + c0],      acc0[r]);
        atomicAdd(&stok[(size_t)bh * 4096 + (gbase + r) * 64 + c0 + 16], acc1[r]);
    }
}

// ---------------- K4: tiny slice attention per (b,h) ----------------
__global__ __launch_bounds__(256) void k4_attn(const float* __restrict__ stok,
                                               const float* __restrict__ norm_g,
                                               const float* __restrict__ Wq,
                                               const float* __restrict__ Wk,
                                               const float* __restrict__ Wv,
                                               _Float16* __restrict__ outs) {
    __shared__ float st[64 * 65];
    __shared__ float qm[64 * 65];
    __shared__ float km[64 * 65];
    __shared__ float vm[64 * 65];
    __shared__ float am[64 * 65];
    const int tid = threadIdx.x, bh = blockIdx.x;
    const int g = tid >> 2, c16 = (tid & 3) * 16;
    {
        float inv = 1.0f / (norm_g[bh * 64 + g] + 1e-5f);
        for (int i = 0; i < 16; i++) {
            int c = c16 + i;
            st[g * 65 + c] = stok[(size_t)bh * 4096 + g * 64 + c] * inv;
        }
    }
    __syncthreads();
    for (int i = 0; i < 16; i++) {
        int c = c16 + i;
        float sq = 0, sk = 0, sv = 0;
        for (int kk = 0; kk < 64; kk++) {
            float s = st[g * 65 + kk];
            sq += s * Wq[c * 64 + kk];
            sk += s * Wk[c * 64 + kk];
            sv += s * Wv[c * 64 + kk];
        }
        qm[g * 65 + c] = sq; km[g * 65 + c] = sk; vm[g * 65 + c] = sv;
    }
    __syncthreads();
    for (int i = 0; i < 16; i++) {
        int j = c16 + i;
        float s = 0;
        for (int kk = 0; kk < 64; kk++) s += qm[g * 65 + kk] * km[j * 65 + kk];
        am[g * 65 + j] = s * 0.125f;   // 1/sqrt(64)
    }
    __syncthreads();
    if (tid < 64) {
        int gr = tid;
        float mx = -1e30f;
        for (int j = 0; j < 64; j++) mx = fmaxf(mx, am[gr * 65 + j]);
        float ssum = 0;
        for (int j = 0; j < 64; j++) { float e = expf(am[gr * 65 + j] - mx); am[gr * 65 + j] = e; ssum += e; }
        float inv = 1.0f / ssum;
        for (int j = 0; j < 64; j++) am[gr * 65 + j] *= inv;
    }
    __syncthreads();
    for (int i = 0; i < 16; i++) {
        int c = c16 + i;
        float s = 0;
        for (int j = 0; j < 64; j++) s += am[g * 65 + j] * vm[j * 65 + c];
        outs[(size_t)bh * 4096 + g * 64 + c] = (_Float16)s;
    }
}

// ---------------- K5: out_token[n, h*64+c] = sum_g slice_w[n,g] * out_s[g,c] ----------------
// 64-token block tile, 8 waves, 2 N-tiles per wave; slice_w staged by async-LDS copy.
__global__ __launch_bounds__(256) void k5_scatter(const _Float16* __restrict__ sw,
                                                  const _Float16* __restrict__ outs,
                                                  _Float16* __restrict__ ot) {
    __shared__ alignas(16) _Float16 Asw[64 * 72];   // [tok][g]
    __shared__ alignas(16) _Float16 Bts[64 * 72];   // [c][g]
    const int tid = threadIdx.x, lane = tid & 31, wid = tid >> 5;
    const int tchunk = blockIdx.x, bh = blockIdx.y;
    const int b_ = bh >> 3, h = bh & 7;
    const int tok0 = tchunk * 64;
    {   // async-stage slice_w tile 64x64 (straight f16 copy, 2x16B per lane)
        int r = tid >> 2, g16 = (tid & 3) * 16;
        const _Float16* src = sw + ((size_t)bh * Ntok + tok0 + r) * Gs + g16;
        async_ld_b128(Asw + r * 72 + g16,     src);
        async_ld_b128(Asw + r * 72 + g16 + 8, src + 8);
    }
    {   // transpose-stage out_s: Bts[c][g]
        int c = tid >> 2, g16 = (tid & 3) * 16;
#pragma unroll
        for (int i = 0; i < 16; i++)
            Bts[c * 72 + g16 + i] = outs[(size_t)bh * 4096 + (size_t)(g16 + i) * 64 + c];
    }
    wait_async0();
    __syncthreads();
    const int wm = wid & 3, wn0 = (wid >> 2) << 1;
    v8f acc0 = {}, acc1 = {};
#pragma unroll
    for (int ks = 0; ks < 64; ks += 32) {
        v16h a  = frag_a(Asw + wm * 16 * 72 + ks, lane, 72);
        v16h b0 = frag_b(Bts + wn0 * 16 * 72 + ks, lane, 72);
        v16h b1 = frag_b(Bts + (wn0 + 1) * 16 * 72 + ks, lane, 72);
        acc0 = WMMA_F16(a, b0, acc0);
        acc1 = WMMA_F16(a, b1, acc1);
    }
    const int mrow = wm * 16 + ((lane >> 4) << 3);
    const int c0 = wn0 * 16 + (lane & 15);
    const int c1 = c0 + 16;
#pragma unroll
    for (int r = 0; r < 8; r++) {
        int tok = tok0 + mrow + r;
        ot[((size_t)b_ * Ntok + tok) * INNER + h * 64 + c0] = (_Float16)acc0[r];
        ot[((size_t)b_ * Ntok + tok) * INNER + h * 64 + c1] = (_Float16)acc1[r];
    }
}

// ---------------- K6: out = out_token @ Wo^T + bo   (M=65536, N=256, K=512) ----------------
// 64x64 block tile, 8 waves, 2 N-tiles per wave; A (f16) staged by async-LDS copy.
__global__ __launch_bounds__(256) void k6_out(const _Float16* __restrict__ ot,
                                              const float* __restrict__ Wo,
                                              const float* __restrict__ bo,
                                              float* __restrict__ out) {
    __shared__ alignas(16) _Float16 As[64 * 40];
    __shared__ alignas(16) _Float16 Bs[64 * 40];
    const int tid = threadIdx.x, lane = tid & 31, wid = tid >> 5;
    const int mblk = blockIdx.x, nblk = blockIdx.y;
    const int wm = wid & 3, wn0 = (wid >> 2) << 1;
    const int sr = tid >> 2, sc8 = (tid & 3) * 8;
    v8f acc0 = {}, acc1 = {};
    for (int k0 = 0; k0 < INNER; k0 += 32) {
        {   // async-stage A tile 64x32 (f16 direct, 16B per lane)
            const _Float16* src = ot + (size_t)(mblk * 64 + sr) * INNER + k0 + sc8;
            async_ld_b128(As + sr * 40 + sc8, src);
            if (k0 + 32 < INNER) __builtin_prefetch(src + 32, 0, 0);
        }
        {   // stage B^T tile 64x32 (f32 -> f16)
            const float* src = Wo + (size_t)(nblk * 64 + sr) * INNER + k0 + sc8;
            float4 v0 = *(const float4*)(src);
            float4 v1 = *(const float4*)(src + 4);
            _Float16* p = Bs + sr * 40 + sc8;
            p[0] = (_Float16)v0.x; p[1] = (_Float16)v0.y; p[2] = (_Float16)v0.z; p[3] = (_Float16)v0.w;
            p[4] = (_Float16)v1.x; p[5] = (_Float16)v1.y; p[6] = (_Float16)v1.z; p[7] = (_Float16)v1.w;
            if (k0 + 32 < INNER) __builtin_prefetch(src + 32, 0, 0);
        }
        wait_async0();
        __syncthreads();
        v16h a  = frag_a(As + wm * 16 * 40, lane, 40);
        v16h b0 = frag_b(Bs + wn0 * 16 * 40, lane, 40);
        v16h b1 = frag_b(Bs + (wn0 + 1) * 16 * 40, lane, 40);
        acc0 = WMMA_F16(a, b0, acc0);
        acc1 = WMMA_F16(a, b1, acc1);
        __syncthreads();
    }
    const int mbase = mblk * 64 + wm * 16 + ((lane >> 4) << 3);
    const int n0 = nblk * 64 + wn0 * 16 + (lane & 15);
    const int n1 = n0 + 16;
    const float bn0 = bo[n0], bn1 = bo[n1];
#pragma unroll
    for (int r = 0; r < 8; r++) {
        out[(size_t)(mbase + r) * Dm + n0] = acc0[r] + bn0;
        out[(size_t)(mbase + r) * Dm + n1] = acc1[r] + bn1;
    }
}

extern "C" void kernel_launch(void* const* d_in, const int* in_sizes, int n_in,
                              void* d_out, int out_size, void* d_ws, size_t ws_size,
                              hipStream_t stream) {
    const float* x   = (const float*)d_in[0];
    const float* Wx  = (const float*)d_in[1];
    const float* bx  = (const float*)d_in[2];
    const float* W1  = (const float*)d_in[3];
    const float* b1  = (const float*)d_in[4];
    const float* W2  = (const float*)d_in[5];
    const float* b2  = (const float*)d_in[6];
    const float* bia = (const float*)d_in[7];
    const float* Wsl = (const float*)d_in[8];
    const float* bsl = (const float*)d_in[9];
    const float* Wq  = (const float*)d_in[10];
    const float* Wk  = (const float*)d_in[11];
    const float* Wv  = (const float*)d_in[12];
    const float* Wo  = (const float*)d_in[13];
    const float* bo  = (const float*)d_in[14];
    const float* u   = (const float*)d_in[15];
    float* out = (float*)d_out;

    char* ws = (char*)d_ws;
    const size_t n_xmid = (size_t)Bb * Hh * Ntok * DH;           // 33.5M elements
    _Float16* xmid  = (_Float16*)(ws);                           // 64 MB (f16)
    _Float16* sw    = (_Float16*)(ws + n_xmid * 2);              // 64 MB (f16)
    float*    normg = (float*)(ws + n_xmid * 4);                 // 8 KB
    float*    stok  = (float*)(ws + n_xmid * 4 + 8192);          // 512 KB
    _Float16* outs  = (_Float16*)(ws + n_xmid * 4 + 8192 + 512 * 1024);  // 256 KB
    _Float16* ot    = xmid;   // out_token aliases x_mid (x_mid dead after k3)

    hipMemsetAsync(normg, 0, 8192 + 512 * 1024, stream);   // zero norm + slice_tok accumulators

    k1_proj_x <<<dim3(1024, 8), 256, 0, stream>>>(x, Wx, bx, xmid);
    k2_route  <<<dim3(512, 32), 256, 0, stream>>>(xmid, W1, b1, W2, b2, bia, Wsl, bsl, u, sw, normg);
    k3_pool   <<<dim3(32, 32),  256, 0, stream>>>(sw, xmid, stok);
    k4_attn   <<<32,            256, 0, stream>>>(stok, normg, Wq, Wk, Wv, outs);
    k5_scatter<<<dim3(256, 32), 256, 0, stream>>>(sw, outs, ot);
    k6_out    <<<dim3(1024, 4), 256, 0, stream>>>(ot, Wo, bo, out);
}
